// RAE_57080115363896
// MI455X (gfx1250) — matrix operand order repaired and checked by reading, not compiled, not run on hardware
//
#include <hip/hip_runtime.h>

// ---------------------------------------------------------------------------
// RAE forward for MI455X (gfx1250): bf16 WMMA GEMMs with async-to-LDS double
// buffering, fused bias/ReLU epilogue, LayerNorm as per-row reduction, pack
// kernels for the concat/ts_enc inputs.
// ---------------------------------------------------------------------------

typedef __attribute__((ext_vector_type(16))) __bf16 v16bf;
typedef __attribute__((ext_vector_type(8)))  float  v8f;

union FragB16 { uint4 u[2]; v16bf v; };

#define BM 128
#define BN 128
#define BK 32

// Async DMA: global memory -> LDS, 16 bytes per lane, tracked by ASYNCcnt.
// vdst VGPR holds the LDS byte address (low 32 bits of the generic pointer,
// per the ISA LDS-aperture rule), vaddr the 64-bit global address.
__device__ __forceinline__ void async_load_b128(unsigned int lds_addr,
                                                const void* gptr)
{
    asm volatile("global_load_async_to_lds_b128 %0, %1, off"
                 :: "v"(lds_addr),
                    "v"((unsigned long long)(uintptr_t)gptr)
                 : "memory");
}

#if __has_builtin(__builtin_amdgcn_s_wait_asynccnt)
#define WAIT_ASYNCCNT(n) __builtin_amdgcn_s_wait_asynccnt(n)
#else
#define WAIT_ASYNCCNT(n) asm volatile("s_wait_asynccnt %0" :: "n"(n) : "memory")
#endif

// A: [M x K] row-major bf16 (K multiple of 32, rows 16B aligned)
// Bt: [N x K] bf16 (weight pre-transposed: Bt[n][k] = W[k][n]); N % 128 == 0
// C: [M x N] fp32, C = relu?(A*B + bias)
__global__ __launch_bounds__(256)
void gemm_bf16(const __bf16* __restrict__ A, const __bf16* __restrict__ Bt,
               const float* __restrict__ bias, float* __restrict__ C,
               int M, int N, int K, int do_relu)
{
    __shared__ __bf16 sA[2][BM][BK];   // 2 x 8 KB, double buffered
    __shared__ __bf16 sB[2][BN][BK];   // 2 x 8 KB (N-major)

    const int tid  = threadIdx.x;
    const int lane = tid & 31;
    const int wave = tid >> 5;
    const int bm   = blockIdx.x * BM;
    const int bn   = blockIdx.y * BN;

    const int wm = (wave & 3) * 32;   // wave's M offset within tile (4 waves)
    const int wn = (wave >> 2) * 64;  // wave's N offset within tile (2 waves)

    v8f acc[2][4];
#pragma unroll
    for (int i = 0; i < 2; ++i)
#pragma unroll
        for (int j = 0; j < 4; ++j)
            acc[i][j] = (v8f){0.f,0.f,0.f,0.f,0.f,0.f,0.f,0.f};

    const int lr = tid >> 1;          // 0..127: tile row this thread stages
    const int lc = (tid & 1) * 16;    // 0 or 16: K sub-segment
    // Clamp OOB A rows instead of skipping: every wave then issues exactly
    // 4 async loads per tile, keeping ASYNCcnt bookkeeping uniform across
    // waves (a branch-skipped wave would pass its wait before its own B-tile
    // loads landed). Duplicate rows only feed C rows >= M, never stored.
    const int ga = (bm + lr < M) ? (bm + lr) : (M - 1);
    const int gb = (bn + lr < N) ? (bn + lr) : (N - 1);

    const int kg  = lane >> 4;        // K group (0/1) per WMMA 16-bit layout
    const int l16 = lane & 15;

    const int ntiles = K / BK;

    auto issue_tile = [&](int t, int buf) {
        const int k0 = t * BK;
        const __bf16* gA = A  + (size_t)ga * K + k0 + lc;
        const __bf16* gB = Bt + (size_t)gb * K + k0 + lc;
        const unsigned int lA =
            (unsigned int)(uintptr_t)&sA[buf][lr][lc];
        const unsigned int lB =
            (unsigned int)(uintptr_t)&sB[buf][lr][lc];
        async_load_b128(lA,      gA);      // K = lc .. lc+7
        async_load_b128(lA + 16, gA + 8);  // K = lc+8 .. lc+15
        async_load_b128(lB,      gB);
        async_load_b128(lB + 16, gB + 8);
    };

    issue_tile(0, 0);
    if (ntiles > 1) issue_tile(1, 1);

    for (int t = 0; t < ntiles; ++t) {
        const int cur = t & 1;
        if (t + 1 < ntiles) WAIT_ASYNCCNT(4);  // next tile may stay in flight
        else                WAIT_ASYNCCNT(0);
        __syncthreads();                       // tile `cur` visible to all waves

        // ---- build fragments in contiguous K order (k = 16*kg + s), same for A and B ----
        // B (32x16): lane = N column; lanes 0-15 hold K 0..15, 16-31 K 16..31.
        FragB16 bfrag[4];
#pragma unroll
        for (int nj = 0; nj < 4; ++nj) {
            const __bf16* src = &sB[cur][wn + nj * 16 + l16][kg * 16];
            bfrag[nj].u[0] = *(const uint4*)(src);
            bfrag[nj].u[1] = *(const uint4*)(src + 8);
        }
        // A (16x32): lane%16 = M row; lanes 0-15 hold K 0..15,
        // lanes 16-31 hold K 16..31.
        FragB16 afrag[2];
#pragma unroll
        for (int mi = 0; mi < 2; ++mi) {
            const __bf16* src = &sA[cur][wm + mi * 16 + l16][kg * 16];
            afrag[mi].u[0] = *(const uint4*)(src);        // K = kg*16 .. +7
            afrag[mi].u[1] = *(const uint4*)(src + 8);    // K = kg*16+8 .. +15
        }

#pragma unroll
        for (int mi = 0; mi < 2; ++mi)
#pragma unroll
            for (int nj = 0; nj < 4; ++nj)
                acc[mi][nj] = __builtin_amdgcn_wmma_f32_16x16x32_bf16(
                    false, afrag[mi].v, false, bfrag[nj].v,
                    (short)0, acc[mi][nj], false, false);

        __syncthreads();                       // everyone done reading `cur`
        if (t + 2 < ntiles) issue_tile(t + 2, cur);
    }

    // ---- epilogue: C/D layout (VGPR r -> M=r / M=8+r by lane half) ----
#pragma unroll
    for (int nj = 0; nj < 4; ++nj) {
        const int col = bn + wn + nj * 16 + l16;
        const float bv = bias[col];
#pragma unroll
        for (int mi = 0; mi < 2; ++mi) {
            const int rbase = bm + wm + mi * 16 + (kg << 3);
#pragma unroll
            for (int r = 0; r < 8; ++r) {
                const int row = rbase + r;
                if (row < M) {
                    float v = acc[mi][nj][r] + bv;
                    if (do_relu) v = fmaxf(v, 0.f);
                    C[(size_t)row * N + col] = v;
                }
            }
        }
    }
}

// ---------------------------------------------------------------------------
// LayerNorm (+ReLU) over last dim: one 256-thread block per row.
__global__ __launch_bounds__(256)
void ln_relu_kernel(const float* __restrict__ in, const float* __restrict__ g,
                    const float* __restrict__ b, float* __restrict__ out, int N)
{
    const int row = blockIdx.x;
    const float* x = in + (size_t)row * N;
    float* y = out + (size_t)row * N;

    float s = 0.f, s2 = 0.f;
    for (int c = threadIdx.x; c < N; c += 256) {
        float v = x[c];
        s += v; s2 += v * v;
    }
    for (int o = 16; o > 0; o >>= 1) {
        s  += __shfl_down(s,  o, 32);
        s2 += __shfl_down(s2, o, 32);
    }
    __shared__ float rs[8], rs2[8];
    const int wave = threadIdx.x >> 5, lane = threadIdx.x & 31;
    if (lane == 0) { rs[wave] = s; rs2[wave] = s2; }
    __syncthreads();
    if (wave == 0) {
        s  = (lane < 8) ? rs[lane]  : 0.f;
        s2 = (lane < 8) ? rs2[lane] : 0.f;
        for (int o = 4; o > 0; o >>= 1) {
            s  += __shfl_down(s,  o, 32);
            s2 += __shfl_down(s2, o, 32);
        }
        if (lane == 0) { rs[0] = s; rs2[0] = s2; }
    }
    __syncthreads();
    const float mean = rs[0] / (float)N;
    const float var  = rs2[0] / (float)N - mean * mean;
    const float inv  = rsqrtf(var + 1e-5f);
    for (int c = threadIdx.x; c < N; c += 256) {
        float v = (x[c] - mean) * inv * g[c] + b[c];
        y[c] = v > 0.f ? v : 0.f;
    }
}

// ---------------------------------------------------------------------------
// fp32 [K,N] weight -> bf16 [N,Kpad] (transposed + K zero-padded)
__global__ void cvt_weight(const float* __restrict__ W, __bf16* __restrict__ Wt,
                           int K, int N, int Kpad)
{
    const size_t total = (size_t)N * Kpad;
    for (size_t i = (size_t)blockIdx.x * 256 + threadIdx.x; i < total;
         i += (size_t)gridDim.x * 256) {
        const int n = (int)(i / Kpad), k = (int)(i % Kpad);
        Wt[i] = (k < K) ? (__bf16)W[(size_t)k * N + n] : (__bf16)0.f;
    }
}

// fp32 [M,K] activations -> bf16 [M,Kpad]
__global__ void cvt_act(const float* __restrict__ src, __bf16* __restrict__ dst,
                        int M, int K, int Kpad)
{
    const size_t total = (size_t)M * Kpad;
    for (size_t i = (size_t)blockIdx.x * 256 + threadIdx.x; i < total;
         i += (size_t)gridDim.x * 256) {
        const size_t r = i / Kpad;
        const int k = (int)(i % Kpad);
        dst[i] = (k < K) ? (__bf16)src[r * K + k] : (__bf16)0.f;
    }
}

__device__ __forceinline__ float ts_val(int rc, int j)
{
    const int idx = (rc == 1) ? 1 : (rc == 2) ? 2 : (rc <= 4) ? 3 : (rc + 10) / 3 - 1;
    if (j == 0) return (float)rc;
    return (j == idx) ? 1.0f : 0.0f;
}

// encode input pack: X [64, Lc+1, 1024] -> A bf16 [64*Lc, 2080]
// row (b,t): [ x[b,t,:] | x[b,t+1,:] | ts_enc(rc) | pad ]
__global__ void pack_enc(const float* __restrict__ X, __bf16* __restrict__ A,
                         int Lc, int rc)
{
    const size_t total = (size_t)64 * Lc * 2080;
    for (size_t i = (size_t)blockIdx.x * 256 + threadIdx.x; i < total;
         i += (size_t)gridDim.x * 256) {
        const int c = (int)(i % 2080);
        const size_t r = i / 2080;
        const int b = (int)(r / Lc), t = (int)(r % Lc);
        float v;
        if (c < 1024)      v = X[((size_t)b * (Lc + 1) + t) * 1024 + c];
        else if (c < 2048) v = X[((size_t)b * (Lc + 1) + t + 1) * 1024 + (c - 1024)];
        else if (c < 2072) v = ts_val(rc, c - 2048);
        else               v = 0.f;
        A[i] = (__bf16)v;
    }
}

// decode input pack: X [64, Li, 1024] -> A bf16 [64*Li, 1056]
__global__ void pack_dec(const float* __restrict__ X, __bf16* __restrict__ A,
                         int Li, int rc)
{
    const size_t total = (size_t)64 * Li * 1056;
    for (size_t i = (size_t)blockIdx.x * 256 + threadIdx.x; i < total;
         i += (size_t)gridDim.x * 256) {
        const int c = (int)(i % 1056);
        const size_t r = i / 1056;
        float v;
        if (c < 1024)      v = X[r * 1024 + c];
        else if (c < 1048) v = ts_val(rc, c - 1024);
        else               v = 0.f;
        A[i] = (__bf16)v;
    }
}

// decode overlap combine: H [64, Lc, 2048] -> Y [64, Lc+1, 1024]
__global__ void dec_combine(const float* __restrict__ H, float* __restrict__ Y,
                            int Lc)
{
    const size_t total = (size_t)64 * (Lc + 1) * 1024;
    for (size_t i = (size_t)blockIdx.x * 256 + threadIdx.x; i < total;
         i += (size_t)gridDim.x * 256) {
        const int c = (int)(i % 1024);
        const size_t r = i / 1024;
        const int b = (int)(r / (Lc + 1)), j = (int)(r % (Lc + 1));
        const float* hb = H + (size_t)b * Lc * 2048;
        float v;
        if (j == 0)       v = hb[c];
        else if (j == Lc) v = hb[(size_t)(Lc - 1) * 2048 + 1024 + c];
        else              v = 0.5f * (hb[(size_t)(j - 1) * 2048 + 1024 + c] +
                                      hb[(size_t)j * 2048 + c]);
        Y[i] = v;
    }
}

// ---------------------------------------------------------------------------
static inline size_t alignup256(size_t x) { return (x + 255) & ~(size_t)255; }

extern "C" void kernel_launch(void* const* d_in, const int* in_sizes, int n_in,
                              void* d_out, int out_size, void* d_ws, size_t ws_size,
                              hipStream_t stream)
{
    (void)in_sizes; (void)n_in; (void)out_size; (void)ws_size;
    const float* x_in    = (const float*)d_in[0];
    const float* se_w1   = (const float*)d_in[1];
    const float* se_b1   = (const float*)d_in[2];
    const float* se_w2   = (const float*)d_in[3];
    const float* se_b2   = (const float*)d_in[4];
    const float* enc_w1  = (const float*)d_in[5];
    const float* enc_b1  = (const float*)d_in[6];
    const float* enc_g1  = (const float*)d_in[7];
    const float* enc_bb1 = (const float*)d_in[8];
    const float* enc_w2  = (const float*)d_in[9];
    const float* enc_b2  = (const float*)d_in[10];
    const float* enc_g2  = (const float*)d_in[11];
    const float* enc_bb2 = (const float*)d_in[12];
    const float* dec_w1  = (const float*)d_in[13];
    const float* dec_b1  = (const float*)d_in[14];
    const float* dec_g1  = (const float*)d_in[15];
    const float* dec_bb1 = (const float*)d_in[16];
    const float* dec_w2  = (const float*)d_in[17];
    const float* dec_b2  = (const float*)d_in[18];
    const float* d2_g    = (const float*)d_in[19];
    const float* d2_b    = (const float*)d_in[20];
    const float* de_w1   = (const float*)d_in[21];
    const float* de_b1   = (const float*)d_in[22];
    const float* de_w2   = (const float*)d_in[23];
    const float* de_b2   = (const float*)d_in[24];

    // ---- workspace layout ----
    char* p = (char*)d_ws;
    auto take = [&](size_t bytes) { void* r = (void*)p; p += alignup256(bytes); return r; };
    __bf16* A   = (__bf16*)take((size_t)4096 * 2080 * 2);   // packed GEMM inputs
    float*  Cb  = (float*) take((size_t)4096 * 2048 * 4);   // GEMM outputs
    float*  X   = (float*) take((size_t)4096 * 1024 * 4);   // current activations
    float*  X2  = (float*) take((size_t)4096 * 1536 * 4);   // intermediates
    __bf16* w_se1t  = (__bf16*)take((size_t)640  * 256  * 2);
    __bf16* w_se2t  = (__bf16*)take((size_t)1024 * 640  * 2);
    __bf16* w_enc1t = (__bf16*)take((size_t)1536 * 2080 * 2);
    __bf16* w_enc2t = (__bf16*)take((size_t)1024 * 1536 * 2);
    __bf16* w_dec1t = (__bf16*)take((size_t)1536 * 1056 * 2);
    __bf16* w_dec2t = (__bf16*)take((size_t)2048 * 1536 * 2);
    __bf16* w_de1t  = (__bf16*)take((size_t)640  * 1024 * 2);
    __bf16* w_de2t  = (__bf16*)take((size_t)256  * 640  * 2);

    const dim3 blk(256);
    auto egrid = [](size_t total) {
        size_t g = (total + 255) / 256;
        if (g > 262144) g = 262144;
        return dim3((unsigned)g);
    };
    auto cvtw = [&](const float* W, __bf16* Wt, int K, int N, int Kpad) {
        cvt_weight<<<egrid((size_t)N * Kpad), blk, 0, stream>>>(W, Wt, K, N, Kpad);
    };
    auto cvta = [&](const float* S, int M, int K, int Kpad) {
        cvt_act<<<egrid((size_t)M * Kpad), blk, 0, stream>>>(S, A, M, K, Kpad);
    };
    auto gemm = [&](const __bf16* Bt, const float* bias, float* Cp,
                    int M, int N, int K, int relu) {
        dim3 grid((M + BM - 1) / BM, N / BN);
        gemm_bf16<<<grid, blk, 0, stream>>>(A, Bt, bias, Cp, M, N, K, relu);
    };
    auto ln = [&](const float* in, const float* g, const float* b, float* out,
                  int M, int N) {
        ln_relu_kernel<<<dim3(M), blk, 0, stream>>>(in, g, b, out, N);
    };

    // ---- weights: fp32 [K,N] -> bf16 [N,Kpad] once per launch ----
    cvtw(se_w1,  w_se1t,  256,  640,  256);
    cvtw(se_w2,  w_se2t,  640,  1024, 640);
    cvtw(enc_w1, w_enc1t, 2072, 1536, 2080);
    cvtw(enc_w2, w_enc2t, 1536, 1024, 1536);
    cvtw(dec_w1, w_dec1t, 1048, 1536, 1056);
    cvtw(dec_w2, w_dec2t, 1536, 2048, 1536);
    cvtw(de_w1,  w_de1t,  1024, 640,  1024);
    cvtw(de_w2,  w_de2t,  640,  256,  640);

    // ---- scale embedding: [4096,256] -> relu -> [4096,640] -> relu -> [4096,1024]
    cvta(x_in, 4096, 256, 256);
    gemm(w_se1t, se_b1, Cb, 4096, 640, 256, 1);
    cvta(Cb, 4096, 640, 640);
    gemm(w_se2t, se_b2, Cb, 4096, 1024, 640, 1);
    hipMemcpyAsync(X, Cb, (size_t)4096 * 1024 * 4, hipMemcpyDeviceToDevice, stream);

    // ---- encode: pairwise merge until length 1 ----
    for (int rc = 1; rc <= 63; ++rc) {
        const int Lc = 64 - rc;
        const int Mr = 64 * Lc;
        pack_enc<<<egrid((size_t)Mr * 2080), blk, 0, stream>>>(X, A, Lc, rc);
        gemm(w_enc1t, enc_b1, Cb, Mr, 1536, 2080, 0);
        ln(Cb, enc_g1, enc_bb1, X2, Mr, 1536);
        cvta(X2, Mr, 1536, 1536);
        gemm(w_enc2t, enc_b2, Cb, Mr, 1024, 1536, 0);
        ln(Cb, enc_g2, enc_bb2, X, Mr, 1024);
    }
    // X = [64, 1, 1024]

    // ---- decode first step: [64,1,1024] -> [64,2,1024] ----
    pack_dec<<<egrid((size_t)64 * 1056), blk, 0, stream>>>(X, A, 1, 63);
    gemm(w_dec1t, dec_b1, Cb, 64, 1536, 1056, 0);
    ln(Cb, dec_g1, dec_bb1, X2, 64, 1536);
    cvta(X2, 64, 1536, 1536);
    gemm(w_dec2t, dec_b2, Cb, 64, 2048, 1536, 0);
    ln(Cb, d2_g, d2_b, X, 128, 1024);   // [64,2048] viewed as [128,1024]

    // ---- decode loop ----
    for (int rc = 62; rc >= 1; --rc) {
        const int Li = 64 - rc;          // input length, output Li+1
        const int Mr = 64 * Li;
        pack_dec<<<egrid((size_t)Mr * 1056), blk, 0, stream>>>(X, A, Li, rc);
        gemm(w_dec1t, dec_b1, Cb, Mr, 1536, 1056, 0);
        ln(Cb, dec_g1, dec_bb1, X2, Mr, 1536);
        cvta(X2, Mr, 1536, 1536);
        gemm(w_dec2t, dec_b2, Cb, Mr, 2048, 1536, 0);
        dec_combine<<<egrid((size_t)64 * (Li + 1) * 1024), blk, 0, stream>>>(Cb, X2, Li);
        ln(X2, d2_g, d2_b, X, 64 * (Li + 1), 1024);
    }
    // X = [64, 64, 1024]

    // ---- descale embedding -> d_out fp32 [64,64,256] ----
    cvta(X, 4096, 1024, 1024);
    gemm(w_de1t, de_b1, Cb, 4096, 640, 1024, 1);
    cvta(Cb, 4096, 640, 640);
    gemm(w_de2t, de_b2, (float*)d_out, 4096, 256, 640, 0);
}
